// PointerNetwork_86749749444873
// MI455X (gfx1250) — matrix operand order repaired and checked by reading, not compiled, not run on hardware
//
#include <hip/hip_runtime.h>
#include <hip/hip_bf16.h>

// Pointer network on MI455X (gfx1250, wave32).
// - Recurrent GEMMs use v_wmma_f32_16x16x32_bf16 (bf16 in, f32 accumulate).
// - One workgroup per 16 batch rows carries (h,c) across all 256 time steps.
// - Wave w owns hidden cols [32w,32w+32): its 8 WMMA tiles are exactly the
//   i/f/g/o gate columns for those j -> LSTM pointwise update stays in regs.
// - Weight B-fragments are re-loaded from global each step (L2-resident,
//   shared by all waves). An opaque per-step BYTE OFFSET (not pointer) stops
//   LICM from hoisting 512 VGPRs of loop-invariant loads into scratch spills
//   while preserving addrspace(1) inference => global_load_b128, not flat.
// - enc_out (128MB f32) is L2-resident; decoder attention context streams it.

#define B_   512
#define S_   256
#define H_   256
#define U_   10
#define G_   1024      // 4*H
#define HPAD 264       // LDS bf16 row stride (16B-aligned, bank-skewed)
#define SLP  260       // LDS f32 row stride for logits / h_f32

typedef __attribute__((ext_vector_type(16))) __bf16 bf16x16;
typedef __attribute__((ext_vector_type(8)))  __bf16 bf16x8;
typedef __attribute__((ext_vector_type(8)))  float  f32x8;

__device__ __forceinline__ __bf16 f2bf(float f) {
  unsigned u = __builtin_bit_cast(unsigned, f);
  unsigned r = u + 0x7FFFu + ((u >> 16) & 1u);   // round-to-nearest-even
  unsigned short s = (unsigned short)(r >> 16);
  return __builtin_bit_cast(__bf16, s);
}
__device__ __forceinline__ float sigm(float x) { return 1.0f / (1.0f + __expf(-x)); }

// Opaque zero: defeats loop-invariant hoisting of weight loads without
// destroying global-addrspace inference on the base pointer.
__device__ __forceinline__ size_t opaque_zero() {
  size_t z = 0;
  asm volatile("" : "+s"(z));
  return z;
}

// A-fragment (16x32 bf16, row-major in LDS): lane m=lane&15, half=lane>>4.
// VGPR0-3 hold K = k0+16*half .. +7 ; VGPR4-7 hold K = k0+16*half+8 .. +15.
// -> two contiguous 16B ds_load_b128 per lane.
__device__ __forceinline__ bf16x16 load_a_frag(const __bf16* base, int row, int k0, int hw) {
  const __bf16* p = base + row * HPAD + k0 + 16 * hw;
  bf16x8 lo = *(const bf16x8*)p;
  bf16x8 hi = *(const bf16x8*)(p + 8);
  return __builtin_shufflevector(lo, hi, 0,1,2,3,4,5,6,7,8,9,10,11,12,13,14,15);
}
// B-fragment (32x16 bf16): lane n=n0+(lane&15); its 16 K-values are
// W[n][k0+16*half .. +15] -> one contiguous 32B global load (weights row-major [G][H]).
__device__ __forceinline__ bf16x16 load_b_frag(const __bf16* W, int n, int k0, int hw) {
  return *(const bf16x16*)(W + (size_t)n * H_ + k0 + 16 * hw);
}

// ---------------- prep: f32 weights -> bf16, combine biases, zero loss ---------------
__global__ __launch_bounds__(256) void prep_kernel(
    const float* __restrict__ encWhh, const float* __restrict__ decWhh,
    const float* __restrict__ decWih,
    const float* __restrict__ ebih, const float* __restrict__ ebhh,
    const float* __restrict__ dbih, const float* __restrict__ dbhh,
    unsigned short* __restrict__ encWhh_bf, unsigned short* __restrict__ decWhh_bf,
    unsigned short* __restrict__ decWihA_bf, float* __restrict__ wih_last,
    float* __restrict__ e_bias, float* __restrict__ d_bias, float* __restrict__ loss)
{
  int idx = blockIdx.x * 256 + threadIdx.x;
  if (idx < G_ * H_) {
    ((__bf16*)encWhh_bf)[idx] = f2bf(encWhh[idx]);
    ((__bf16*)decWhh_bf)[idx] = f2bf(decWhh[idx]);
    int n = idx / H_, k = idx % H_;
    ((__bf16*)decWihA_bf)[idx] = f2bf(decWih[n * (H_ + 1) + k]);
  }
  if (idx < G_) {
    wih_last[idx] = decWih[idx * (H_ + 1) + H_];
    e_bias[idx] = ebih[idx] + ebhh[idx];
    d_bias[idx] = dbih[idx] + dbhh[idx];
  }
  if (idx == 0) *loss = 0.0f;
}

// ---------------- encoder: LSTM over 256 steps, 16 batch rows / workgroup ------------
__global__ __launch_bounds__(256) void enc_kernel(
    const float* __restrict__ x, const float* __restrict__ Wih,
    const unsigned short* __restrict__ Whh_us, const float* __restrict__ bias,
    float* __restrict__ enc_out, float* __restrict__ hN, float* __restrict__ cN)
{
  __shared__ __bf16 s_h[16 * HPAD];
  __shared__ float  s_x[16];

  const int tid  = threadIdx.x;
  const int wave = tid >> 5, lane = tid & 31, hw = lane >> 4, ln = lane & 15;
  const int m0 = blockIdx.x * 16;
  const int jbase = wave * 32;

  for (int i = tid; i < 16 * HPAD; i += 256) s_h[i] = f2bf(0.0f);
  float c[2][8];
  #pragma unroll
  for (int jt = 0; jt < 2; ++jt)
    #pragma unroll
    for (int r = 0; r < 8; ++r) c[jt][r] = 0.0f;

  for (int t = 0; t < S_; ++t) {
    // Per-step opaque offset keeps weight loads inside the t-loop (no 512-VGPR
    // spill) while staying provably global (global_load_b128, LOADcnt only).
    const __bf16* Wt = (const __bf16*)Whh_us + opaque_zero();

    if (tid < 16) s_x[tid] = x[(size_t)(m0 + tid) * S_ + t];
    __syncthreads();

    f32x8 acc[4][2];
    #pragma unroll
    for (int gi = 0; gi < 4; ++gi)
      #pragma unroll
      for (int jt = 0; jt < 2; ++jt) acc[gi][jt] = (f32x8)0.0f;

    for (int kt = 0; kt < 8; ++kt) {
      bf16x16 a = load_a_frag(s_h, ln, kt * 32, hw);
      bf16x16 bfr[8];
      #pragma unroll
      for (int gi = 0; gi < 4; ++gi)
        #pragma unroll
        for (int jt = 0; jt < 2; ++jt)
          bfr[gi * 2 + jt] = load_b_frag(Wt, gi * 256 + jbase + jt * 16 + ln, kt * 32, hw);
      #pragma unroll
      for (int gi = 0; gi < 4; ++gi)
        #pragma unroll
        for (int jt = 0; jt < 2; ++jt)
          acc[gi][jt] = __builtin_amdgcn_wmma_f32_16x16x32_bf16(
              false, a, false, bfr[gi * 2 + jt], (short)0, acc[gi][jt], false, false);
    }

    // pointwise LSTM update, all in registers (thread owns (m=r+8*hw, j))
    float hout[2][8];
    #pragma unroll
    for (int jt = 0; jt < 2; ++jt) {
      int j = jbase + jt * 16 + ln;
      float wi = Wih[j], wf = Wih[256 + j], wg = Wih[512 + j], wo = Wih[768 + j];
      float bi = bias[j], bf = bias[256 + j], bg = bias[512 + j], bo = bias[768 + j];
      #pragma unroll
      for (int r = 0; r < 8; ++r) {
        int m = r + 8 * hw;
        float xt = s_x[m];
        float gi_ = sigm(acc[0][jt][r] + bi + xt * wi);
        float gf_ = sigm(acc[1][jt][r] + bf + xt * wf);
        float gg_ = tanhf(acc[2][jt][r] + bg + xt * wg);
        float go_ = sigm(acc[3][jt][r] + bo + xt * wo);
        float cn = gf_ * c[jt][r] + gi_ * gg_;
        c[jt][r] = cn;
        hout[jt][r] = go_ * tanhf(cn);
      }
    }
    __syncthreads();   // everyone done reading s_h of step t
    #pragma unroll
    for (int jt = 0; jt < 2; ++jt) {
      int j = jbase + jt * 16 + ln;
      #pragma unroll
      for (int r = 0; r < 8; ++r) {
        int m = r + 8 * hw;
        float hv = hout[jt][r];
        s_h[m * HPAD + j] = f2bf(hv);
        enc_out[((size_t)(m0 + m) * S_ + t) * H_ + j] = hv;
        if (t == S_ - 1) {
          hN[(size_t)(m0 + m) * H_ + j] = hv;
          cN[(size_t)(m0 + m) * H_ + j] = c[jt][r];
        }
      }
    }
  }
}

// ---------------- encW1[b,s,u] = enc_out[b,s,:] . W1[u,:] ---------------------------
__global__ __launch_bounds__(256) void encw1_kernel(
    const float* __restrict__ enc_out, const float* __restrict__ W1,
    float* __restrict__ encW1)
{
  __shared__ float s_w1[U_ * H_];
  for (int i = threadIdx.x; i < U_ * H_; i += 256) s_w1[i] = W1[i];
  __syncthreads();
  int gid = blockIdx.x * 256 + threadIdx.x;           // 0 .. B*S-1
  const float* row = enc_out + (size_t)gid * H_;
  float acc[U_];
  #pragma unroll
  for (int u = 0; u < U_; ++u) acc[u] = 0.0f;
  for (int k = 0; k < H_; k += 4) {
    float4 xv = *(const float4*)(row + k);
    #pragma unroll
    for (int u = 0; u < U_; ++u)
      acc[u] += xv.x * s_w1[u * H_ + k] + xv.y * s_w1[u * H_ + k + 1] +
                xv.z * s_w1[u * H_ + k + 2] + xv.w * s_w1[u * H_ + k + 3];
  }
  #pragma unroll
  for (int u = 0; u < U_; ++u) encW1[(size_t)gid * U_ + u] = acc[u];
}

// ---------------- decoder: attention + LSTM, 256 steps ------------------------------
__global__ __launch_bounds__(256) void dec_kernel(
    const float* __restrict__ x, const long long* __restrict__ y,
    const float* __restrict__ encW1, const float* __restrict__ enc_out,
    const float* __restrict__ W2, const float* __restrict__ attV,
    const unsigned short* __restrict__ WihA_us, const unsigned short* __restrict__ Whh_us,
    const float* __restrict__ wih_last, const float* __restrict__ bias,
    const float* __restrict__ hN, const float* __restrict__ cN,
    long long* __restrict__ preds, float* __restrict__ loss)
{
  __shared__ __bf16 s_h[16 * HPAD];
  __shared__ __bf16 s_di[16 * HPAD];
  __shared__ float  s_hf[16 * SLP];
  __shared__ float  s_l[16 * SLP];          // logits, then softmax weights
  __shared__ float  s_w2h[16 * U_];
  __shared__ float  s_pmax[16 * 16], s_psum[16 * 16];
  __shared__ int    s_pidx[16 * 16];
  __shared__ float  s_rowmax[16], s_rowsum[16], s_lossrow[16];
  __shared__ float  s_decin[16];

  const int tid  = threadIdx.x;
  const int wave = tid >> 5, lane = tid & 31, hw = lane >> 4, ln = lane & 15;
  const int m0 = blockIdx.x * 16;
  const int jbase = wave * 32;

  float Vv[U_];
  #pragma unroll
  for (int u = 0; u < U_; ++u) Vv[u] = attV[u];

  float c[2][8];
  #pragma unroll
  for (int jt = 0; jt < 2; ++jt) {
    int j = jbase + jt * 16 + ln;
    #pragma unroll
    for (int r = 0; r < 8; ++r) {
      int m = r + 8 * hw;
      float hv = hN[(size_t)(m0 + m) * H_ + j];
      c[jt][r]  = cN[(size_t)(m0 + m) * H_ + j];
      s_h[m * HPAD + j] = f2bf(hv);
      s_hf[m * SLP + j] = hv;
    }
  }
  if (tid < 16) s_decin[tid] = 0.0f;

  for (int t = 0; t < S_; ++t) {
    // Per-step opaque offsets (see encoder comment).
    const __bf16* WtA = (const __bf16*)WihA_us + opaque_zero();
    const __bf16* WtH = (const __bf16*)Whh_us + opaque_zero();

    __syncthreads();                                  // h / decin of step t ready

    // W2h[m][u] = h[m,:] . W2[u,:]
    if (tid < 16 * U_) {
      int m = tid / U_, u = tid % U_;
      float a = 0.0f;
      for (int k = 0; k < H_; ++k) a += s_hf[m * SLP + k] * W2[u * H_ + k];
      s_w2h[m * U_ + u] = a;
    }
    __syncthreads();

    // logits[m][s] = sum_u V[u] * tanh(encW1[b,s,u] + W2h[m,u])
    {
      int m = tid >> 4, sb = tid & 15;
      for (int it = 0; it < 16; ++it) {
        int s = sb + 16 * it;
        const float* ew = encW1 + ((size_t)(m0 + m) * S_ + s) * U_;
        float lt = 0.0f;
        #pragma unroll
        for (int u = 0; u < U_; ++u) lt += Vv[u] * tanhf(ew[u] + s_w2h[m * U_ + u]);
        s_l[m * SLP + s] = lt;
      }
    }
    __syncthreads();

    // row max + argmax (first-occurrence), 16 threads per row
    {
      int m = tid >> 4, q = tid & 15;
      float mx = -3.4e38f; int idx = 0;
      for (int s = q * 16; s < q * 16 + 16; ++s) {
        float v = s_l[m * SLP + s];
        if (v > mx) { mx = v; idx = s; }
      }
      s_pmax[m * 16 + q] = mx; s_pidx[m * 16 + q] = idx;
    }
    __syncthreads();
    if (tid < 16) {
      int m = tid;
      float mx = -3.4e38f; int idx = 0;
      for (int q = 0; q < 16; ++q) {
        float v = s_pmax[m * 16 + q];
        if (v > mx) { mx = v; idx = s_pidx[m * 16 + q]; }
      }
      s_rowmax[m] = mx;
      preds[(size_t)t * B_ + (m0 + m)] = (long long)idx;
    }
    __syncthreads();

    // row sum of exp
    {
      int m = tid >> 4, q = tid & 15;
      float rm = s_rowmax[m], sm = 0.0f;
      for (int s = q * 16; s < q * 16 + 16; ++s) sm += __expf(s_l[m * SLP + s] - rm);
      s_psum[m * 16 + q] = sm;
    }
    __syncthreads();
    if (tid < 16) {
      int m = tid;
      float sm = 0.0f;
      for (int q = 0; q < 16; ++q) sm += s_psum[m * 16 + q];
      s_rowsum[m] = sm;
      int yt = (int)y[(size_t)(m0 + m) * S_ + t];
      float lg = s_l[m * SLP + yt];
      s_lossrow[m] = -((lg - s_rowmax[m]) - __logf(sm));
    }
    __syncthreads();
    if (tid == 0) {
      float bl = 0.0f;
      for (int m = 0; m < 16; ++m) bl += s_lossrow[m];
      atomicAdd(loss, bl * (1.0f / ((float)B_ * (float)B_)));
    }
    // softmax weights overwrite logits
    {
      int m = tid >> 4, sb = tid & 15;
      float rm = s_rowmax[m], inv = 1.0f / s_rowsum[m];
      for (int it = 0; it < 16; ++it) {
        int s = sb + 16 * it;
        s_l[m * SLP + s] = __expf(s_l[m * SLP + s] - rm) * inv;
      }
    }
    __syncthreads();

    // di[m][:] = sum_s aj[m][s] * enc_out[b,s,:]  (L2-resident stream)
    {
      int m = tid >> 4, hb = (tid & 15) * 16;
      float a16[16];
      #pragma unroll
      for (int i = 0; i < 16; ++i) a16[i] = 0.0f;
      const float* ebase = enc_out + (size_t)(m0 + m) * S_ * H_ + hb;
      for (int s = 0; s < S_; ++s) {
        float a = s_l[m * SLP + s];
        const float4* e = (const float4*)(ebase + (size_t)s * H_);
        #pragma unroll
        for (int q = 0; q < 4; ++q) {
          float4 ev = e[q];
          a16[q * 4 + 0] += a * ev.x; a16[q * 4 + 1] += a * ev.y;
          a16[q * 4 + 2] += a * ev.z; a16[q * 4 + 3] += a * ev.w;
        }
      }
      #pragma unroll
      for (int i = 0; i < 16; ++i) s_di[m * HPAD + hb + i] = f2bf(a16[i]);
    }
    __syncthreads();

    // gates = di @ WihA^T + h @ Whh^T   (WMMA, f32 accumulate)
    f32x8 acc[4][2];
    #pragma unroll
    for (int gi = 0; gi < 4; ++gi)
      #pragma unroll
      for (int jt = 0; jt < 2; ++jt) acc[gi][jt] = (f32x8)0.0f;

    for (int kt = 0; kt < 8; ++kt) {
      bf16x16 ad = load_a_frag(s_di, ln, kt * 32, hw);
      bf16x16 ah = load_a_frag(s_h,  ln, kt * 32, hw);
      bf16x16 bfr[8];
      // pass 1: di @ WihA^T
      #pragma unroll
      for (int gi = 0; gi < 4; ++gi)
        #pragma unroll
        for (int jt = 0; jt < 2; ++jt)
          bfr[gi * 2 + jt] = load_b_frag(WtA, gi * 256 + jbase + jt * 16 + ln, kt * 32, hw);
      #pragma unroll
      for (int gi = 0; gi < 4; ++gi)
        #pragma unroll
        for (int jt = 0; jt < 2; ++jt)
          acc[gi][jt] = __builtin_amdgcn_wmma_f32_16x16x32_bf16(
              false, ad, false, bfr[gi * 2 + jt], (short)0, acc[gi][jt], false, false);
      // pass 2: h @ Whh^T
      #pragma unroll
      for (int gi = 0; gi < 4; ++gi)
        #pragma unroll
        for (int jt = 0; jt < 2; ++jt)
          bfr[gi * 2 + jt] = load_b_frag(WtH, gi * 256 + jbase + jt * 16 + ln, kt * 32, hw);
      #pragma unroll
      for (int gi = 0; gi < 4; ++gi)
        #pragma unroll
        for (int jt = 0; jt < 2; ++jt)
          acc[gi][jt] = __builtin_amdgcn_wmma_f32_16x16x32_bf16(
              false, ah, false, bfr[gi * 2 + jt], (short)0, acc[gi][jt], false, false);
    }

    float hout[2][8];
    #pragma unroll
    for (int jt = 0; jt < 2; ++jt) {
      int j = jbase + jt * 16 + ln;
      float wi = wih_last[j], wf = wih_last[256 + j], wg = wih_last[512 + j], wo = wih_last[768 + j];
      float bi = bias[j], bf = bias[256 + j], bg = bias[512 + j], bo = bias[768 + j];
      #pragma unroll
      for (int r = 0; r < 8; ++r) {
        int m = r + 8 * hw;
        float din = s_decin[m];
        float gi_ = sigm(acc[0][jt][r] + bi + din * wi);
        float gf_ = sigm(acc[1][jt][r] + bf + din * wf);
        float gg_ = tanhf(acc[2][jt][r] + bg + din * wg);
        float go_ = sigm(acc[3][jt][r] + bo + din * wo);
        float cn = gf_ * c[jt][r] + gi_ * gg_;
        c[jt][r] = cn;
        hout[jt][r] = go_ * tanhf(cn);
      }
    }
    __syncthreads();                                  // all GEMM reads of s_h done
    #pragma unroll
    for (int jt = 0; jt < 2; ++jt) {
      int j = jbase + jt * 16 + ln;
      #pragma unroll
      for (int r = 0; r < 8; ++r) {
        int m = r + 8 * hw;
        float hv = hout[jt][r];
        s_h[m * HPAD + j] = f2bf(hv);
        s_hf[m * SLP + j] = hv;
      }
    }
    if (tid < 16) {                                   // teacher forcing input for t+1
      int yt = (int)y[(size_t)(m0 + tid) * S_ + t];
      s_decin[tid] = x[(size_t)(m0 + tid) * S_ + yt];
    }
  }
}

// ---------------- host launcher ------------------------------------------------------
extern "C" void kernel_launch(void* const* d_in, const int* in_sizes, int n_in,
                              void* d_out, int out_size, void* d_ws, size_t ws_size,
                              hipStream_t stream) {
  const float*     x       = (const float*)d_in[0];
  const long long* y       = (const long long*)d_in[1];
  const float*     enc_Wih = (const float*)d_in[2];
  const float*     enc_Whh = (const float*)d_in[3];
  const float*     enc_bih = (const float*)d_in[4];
  const float*     enc_bhh = (const float*)d_in[5];
  const float*     dec_Wih = (const float*)d_in[6];
  const float*     dec_Whh = (const float*)d_in[7];
  const float*     dec_bih = (const float*)d_in[8];
  const float*     dec_bhh = (const float*)d_in[9];
  const float*     att_W1  = (const float*)d_in[10];
  const float*     att_W2  = (const float*)d_in[11];
  const float*     att_V   = (const float*)d_in[12];

  char* ws = (char*)d_ws;
  size_t off = 0;
  auto carve = [&](size_t bytes) -> void* {
    void* p = ws + off;
    off += (bytes + 255) & ~(size_t)255;
    return p;
  };
  float*          enc_out    = (float*)carve((size_t)B_ * S_ * H_ * 4);   // 128 MB (L2-resident)
  float*          encW1      = (float*)carve((size_t)B_ * S_ * U_ * 4);   // 5 MB
  float*          hN         = (float*)carve((size_t)B_ * H_ * 4);
  float*          cN         = (float*)carve((size_t)B_ * H_ * 4);
  unsigned short* encWhh_bf  = (unsigned short*)carve((size_t)G_ * H_ * 2);
  unsigned short* decWhh_bf  = (unsigned short*)carve((size_t)G_ * H_ * 2);
  unsigned short* decWihA_bf = (unsigned short*)carve((size_t)G_ * H_ * 2);
  float*          wih_last   = (float*)carve((size_t)G_ * 4);
  float*          e_bias     = (float*)carve((size_t)G_ * 4);
  float*          d_bias     = (float*)carve((size_t)G_ * 4);

  long long* preds = (long long*)d_out;                          // [S, B, 1] int64
  float*     loss  = (float*)(preds + (size_t)S_ * B_);          // scalar after preds

  prep_kernel<<<(G_ * H_ + 255) / 256, 256, 0, stream>>>(
      enc_Whh, dec_Whh, dec_Wih, enc_bih, enc_bhh, dec_bih, dec_bhh,
      encWhh_bf, decWhh_bf, decWihA_bf, wih_last, e_bias, d_bias, loss);

  enc_kernel<<<B_ / 16, 256, 0, stream>>>(
      x, enc_Wih, encWhh_bf, e_bias, enc_out, hN, cN);

  encw1_kernel<<<(B_ * S_) / 256, 256, 0, stream>>>(enc_out, att_W1, encW1);

  dec_kernel<<<B_ / 16, 256, 0, stream>>>(
      x, y, encW1, enc_out, att_W2, att_V,
      decWihA_bf, decWhh_bf, wih_last, d_bias, hN, cN, preds, loss);
}
